// PartialAttention_30812095381697
// MI455X (gfx1250) — matrix operand order repaired and checked by reading, not compiled, not run on hardware
//
#include <hip/hip_runtime.h>
#include <hip/hip_bf16.h>
#include <math.h>

#define SS 4096
#define BB 2
#define EE 1024
#define DD 64

typedef __attribute__((ext_vector_type(16))) _Float16 v16h;
typedef __attribute__((ext_vector_type(8)))  _Float16 v8h;
typedef __attribute__((ext_vector_type(8)))  float    v8f;

struct alignas(8) h4 { _Float16 h[4]; };

static __device__ __forceinline__ v16h cat16(v8h lo, v8h hi) {
  v16h r;
#pragma unroll
  for (int i = 0; i < 8; ++i) { r[i] = lo[i]; r[i + 8] = hi[i]; }
  return r;
}

// ---------------------------------------------------------------------------
// Kernel W: build WhT[128][1024] f16 where col c<64 -> Wq[:,c], c>=64 -> Wk.
// Column-major-by-output so WMMA B-fragments are contiguous along K (=E).
// ---------------------------------------------------------------------------
__global__ __launch_bounds__(256) void prep_w_kernel(
    const float* __restrict__ Wq, const float* __restrict__ Wk,
    _Float16* __restrict__ WhT) {
  int t = blockIdx.x * 256 + threadIdx.x;   // 0 .. 131071
  int c = t >> 10;
  int e = t & 1023;
  float v = (c < DD) ? Wq[e * DD + c] : Wk[e * DD + (c - DD)];
  WhT[(size_t)c * EE + e] = (_Float16)v;
}

// ---------------------------------------------------------------------------
// Kernel P: LayerNorm (fp32, wave32 shuffle reductions) + fused Q/K projection
// via v_wmma_f32_16x16x32_f16.  32 rows per workgroup, 8 waves.
//   M tiles: 2 (wave&1), N tiles: 8 (2 per wave).
// Q stored pre-scaled by 1/sqrt(D); biases added.
// ---------------------------------------------------------------------------
__global__ __launch_bounds__(256) void ln_proj_kernel(
    const float* __restrict__ x,       // [S][B][E]
    const float* __restrict__ gamma,
    const float* __restrict__ beta,
    const float* __restrict__ bq,
    const float* __restrict__ bk,
    const _Float16* __restrict__ WhT,  // [128][E]
    _Float16* __restrict__ Qbuf,       // [B*S][64] (scaled)
    _Float16* __restrict__ Kbuf) {     // [B*S][64]
  __shared__ _Float16 xn[32][EE];      // 64 KB normalized-row tile

  const int tid  = threadIdx.x;
  const int wave = tid >> 5;
  const int lane = tid & 31;
  const int wg   = blockIdx.x;         // 0..255, 32 rows each

  // ---- LayerNorm phase: each wave handles 4 rows ----
  for (int rr = 0; rr < 4; ++rr) {
    const int lrow = wave * 4 + rr;
    const int gr   = wg * 32 + lrow;          // flattened (b,s)
    const int b    = gr >> 12;
    const int s    = gr & (SS - 1);
    const float* xrow = x + ((size_t)s * BB + b) * EE;

    float4 xv[8];
    float s1 = 0.f, s2 = 0.f;
#pragma unroll
    for (int j = 0; j < 8; ++j) {
      xv[j] = *(const float4*)(xrow + j * 128 + lane * 4);
      s1 += xv[j].x + xv[j].y + xv[j].z + xv[j].w;
      s2 += xv[j].x * xv[j].x + xv[j].y * xv[j].y +
            xv[j].z * xv[j].z + xv[j].w * xv[j].w;
    }
#pragma unroll
    for (int m = 1; m < 32; m <<= 1) {
      s1 += __shfl_xor(s1, m, 32);
      s2 += __shfl_xor(s2, m, 32);
    }
    const float mu   = s1 * (1.0f / EE);
    const float var  = s2 * (1.0f / EE) - mu * mu;
    const float rstd = rsqrtf(var + 1e-5f);

#pragma unroll
    for (int j = 0; j < 8; ++j) {
      const int e = j * 128 + lane * 4;
      const float4 g  = *(const float4*)(gamma + e);
      const float4 bt = *(const float4*)(beta + e);
      h4 o;
      o.h[0] = (_Float16)((xv[j].x - mu) * rstd * g.x + bt.x);
      o.h[1] = (_Float16)((xv[j].y - mu) * rstd * g.y + bt.y);
      o.h[2] = (_Float16)((xv[j].z - mu) * rstd * g.z + bt.z);
      o.h[3] = (_Float16)((xv[j].w - mu) * rstd * g.w + bt.w);
      *(h4*)(&xn[lrow][e]) = o;
    }
  }
  __syncthreads();

  // ---- WMMA projection phase ----
  const int mt  = wave & 1;           // row tile 0/1
  const int nt0 = (wave >> 1) * 2;    // first of 2 col tiles
  const int hl  = lane & 15;
  const int hi  = lane >> 4;

  v8f acc[2] = {};
  for (int kb = 0; kb < EE; kb += 32) {
    const int arow = mt * 16 + hl;
    const int kofs = hi ? 16 : 0;
    v8h alo = *(const v8h*)(&xn[arow][kb + kofs]);
    v8h ahi = *(const v8h*)(&xn[arow][kb + 8 + kofs]);
    v16h afrag = cat16(alo, ahi);
#pragma unroll
    for (int n = 0; n < 2; ++n) {
      const int col = (nt0 + n) * 16 + hl;
      v16h bfrag = *(const v16h*)(WhT + (size_t)col * EE + kb + (hi ? 16 : 0));
      acc[n] = __builtin_amdgcn_wmma_f32_16x16x32_f16(
          false, afrag, false, bfrag, (short)0, acc[n], false, false);
    }
  }

  // ---- epilogue: bias, scale, store f16 Q/K ----
#pragma unroll
  for (int n = 0; n < 2; ++n) {
    const int col = (nt0 + n) * 16 + hl;
#pragma unroll
    for (int r = 0; r < 8; ++r) {
      const int lrow = mt * 16 + (hi ? 8 : 0) + r;
      const int gr   = wg * 32 + lrow;
      float v = acc[n][r];
      if (col < DD) {
        v = (v + bq[col]) * 0.125f;                 // 1/sqrt(64)
        Qbuf[(size_t)gr * DD + col] = (_Float16)v;
      } else {
        v = v + bk[col - DD];
        Kbuf[(size_t)gr * DD + (col - DD)] = (_Float16)v;
      }
    }
  }
}

// ---------------------------------------------------------------------------
// Kernel S: scores = exp(Q·K^T - rowmax).  One WG = 32 query rows x all 4096
// keys, 4 waves.  Two-pass: pass 1 WMMA + running row-max (registers ->
// xor-shuffle -> LDS across waves), pass 2 recompute WMMA, exp, store fp32.
// ---------------------------------------------------------------------------
__global__ __launch_bounds__(128) void scores_kernel(
    const _Float16* __restrict__ Qbuf,   // [B*S][64] scaled
    const _Float16* __restrict__ Kbuf,   // [B*S][64]
    float* __restrict__ out) {           // [B][S][S]
  __shared__ _Float16 qtile[32][DD];     // 4 KB
  __shared__ float wavemax[4][32];

  const int tid  = threadIdx.x;
  const int wave = tid >> 5;
  const int lane = tid & 31;
  const int wg   = blockIdx.x;           // B * 128
  const int b    = wg >> 7;
  const int rt   = wg & 127;
  const size_t rowbase = (size_t)b * SS + (size_t)rt * 32;

  // load the 32x64 Q stripe
  {
    const int r   = tid >> 2;
    const int dof = (tid & 3) * 16;
    *(v16h*)(&qtile[r][dof]) = *(const v16h*)(Qbuf + (rowbase + r) * DD + dof);
  }
  __syncthreads();

  const int hl = lane & 15;
  const int hi = lane >> 4;

  // A fragments pinned in registers: 2 row tiles x 2 K chunks
  v16h afrag[2][2];
#pragma unroll
  for (int mt = 0; mt < 2; ++mt)
#pragma unroll
    for (int kc = 0; kc < 2; ++kc) {
      const int arow = mt * 16 + hl;
      const int kofs = kc * 32 + (hi ? 16 : 0);
      v8h lo = *(const v8h*)(&qtile[arow][kofs]);
      v8h hh = *(const v8h*)(&qtile[arow][kofs + 8]);
      afrag[mt][kc] = cat16(lo, hh);
    }

  const _Float16* kbase = Kbuf + (size_t)b * SS * DD;

  float runmax[2][8];
#pragma unroll
  for (int mt = 0; mt < 2; ++mt)
#pragma unroll
    for (int r = 0; r < 8; ++r) runmax[mt][r] = -INFINITY;

  // ---- pass 1: row maxima ----
  for (int ct = wave; ct < SS / 16; ct += 4) {
    const int key = ct * 16 + hl;
    const _Float16* kp = kbase + (size_t)key * DD + (hi ? 16 : 0);
    v16h b0 = *(const v16h*)kp;
    v16h b1 = *(const v16h*)(kp + 32);
#pragma unroll
    for (int mt = 0; mt < 2; ++mt) {
      v8f acc = {};
      acc = __builtin_amdgcn_wmma_f32_16x16x32_f16(
          false, afrag[mt][0], false, b0, (short)0, acc, false, false);
      acc = __builtin_amdgcn_wmma_f32_16x16x32_f16(
          false, afrag[mt][1], false, b1, (short)0, acc, false, false);
#pragma unroll
      for (int r = 0; r < 8; ++r)
        runmax[mt][r] = fmaxf(runmax[mt][r], acc[r]);
    }
  }
  // reduce across the 16 lanes sharing each row (xor masks 1,2,4,8)
#pragma unroll
  for (int m = 1; m < 16; m <<= 1)
#pragma unroll
    for (int mt = 0; mt < 2; ++mt)
#pragma unroll
      for (int r = 0; r < 8; ++r)
        runmax[mt][r] = fmaxf(runmax[mt][r], __shfl_xor(runmax[mt][r], m, 32));
  if (hl == 0) {
#pragma unroll
    for (int mt = 0; mt < 2; ++mt)
#pragma unroll
      for (int r = 0; r < 8; ++r)
        wavemax[wave][mt * 16 + hi * 8 + r] = runmax[mt][r];
  }
  __syncthreads();

  float rm[2][8];
#pragma unroll
  for (int mt = 0; mt < 2; ++mt)
#pragma unroll
    for (int r = 0; r < 8; ++r) {
      const int row = mt * 16 + hi * 8 + r;
      float v = wavemax[0][row];
      v = fmaxf(v, wavemax[1][row]);
      v = fmaxf(v, wavemax[2][row]);
      v = fmaxf(v, wavemax[3][row]);
      rm[mt][r] = v;
    }

  // ---- pass 2: recompute, exp, store ----
  for (int ct = wave; ct < SS / 16; ct += 4) {
    const int key = ct * 16 + hl;
    const _Float16* kp = kbase + (size_t)key * DD + (hi ? 16 : 0);
    v16h b0 = *(const v16h*)kp;
    v16h b1 = *(const v16h*)(kp + 32);
#pragma unroll
    for (int mt = 0; mt < 2; ++mt) {
      v8f acc = {};
      acc = __builtin_amdgcn_wmma_f32_16x16x32_f16(
          false, afrag[mt][0], false, b0, (short)0, acc, false, false);
      acc = __builtin_amdgcn_wmma_f32_16x16x32_f16(
          false, afrag[mt][1], false, b1, (short)0, acc, false, false);
#pragma unroll
      for (int r = 0; r < 8; ++r) {
        const float v = __expf(acc[r] - rm[mt][r]);
        const size_t row = rowbase + mt * 16 + hi * 8 + r;
        const int col = ct * 16 + hl;
        out[row * SS + col] = v;
      }
    }
  }
}

// ---------------------------------------------------------------------------
extern "C" void kernel_launch(void* const* d_in, const int* in_sizes, int n_in,
                              void* d_out, int out_size, void* d_ws, size_t ws_size,
                              hipStream_t stream) {
  const float* src_emb = (const float*)d_in[0];
  const float* gamma   = (const float*)d_in[1];
  const float* beta    = (const float*)d_in[2];
  const float* Wq      = (const float*)d_in[3];
  const float* bq      = (const float*)d_in[4];
  const float* Wk      = (const float*)d_in[5];
  const float* bk      = (const float*)d_in[6];
  float* out = (float*)d_out;

  // workspace carve-out (all 256B-aligned offsets)
  char* ws = (char*)d_ws;
  _Float16* WhT  = (_Float16*)(ws);                       // 128*1024*2 = 256 KB
  _Float16* Qbuf = (_Float16*)(ws + (256u << 10));        // 1 MB
  _Float16* Kbuf = (_Float16*)(ws + (256u << 10) + (1u << 20)); // 1 MB

  prep_w_kernel<<<(128 * EE) / 256, 256, 0, stream>>>(Wq, Wk, WhT);
  ln_proj_kernel<<<(BB * SS) / 32, 256, 0, stream>>>(
      src_emb, gamma, beta, bq, bk, WhT, Qbuf, Kbuf);
  scores_kernel<<<BB * (SS / 32), 128, 0, stream>>>(Qbuf, Kbuf, out);
}
